// TritterAttention_19653770346747
// MI455X (gfx1250) — matrix operand order repaired and checked by reading, not compiled, not run on hardware
//
#include <hip/hip_runtime.h>

// ---------------------------------------------------------------------------
// Problem constants (from reference)
// ---------------------------------------------------------------------------
#define S    2048
#define HID  4096
#define NH   32
#define NKV  8
#define HD   128
#define DQ   (NH * HD)    // 4096
#define DKV  (NKV * HD)   // 1024
#define EPS  1e-5f
#define SOFTMAX_SCALE 0.08838834764831845f   // 1/sqrt(128)
#define LOG_ROPE_THETA 13.122363377404328f   // ln(500000)

typedef __bf16 bf16;
typedef bf16  v4bf  __attribute__((ext_vector_type(4)));
typedef bf16  v8bf  __attribute__((ext_vector_type(8)));
typedef bf16  v16bf __attribute__((ext_vector_type(16)));
typedef float v8f   __attribute__((ext_vector_type(8)));

__device__ __forceinline__ bf16 f2bf(float f) { return (bf16)f; }

__device__ __forceinline__ v16bf cat8(v8bf lo, v8bf hi) {
  return __builtin_shufflevector(lo, hi, 0,1,2,3,4,5,6,7,8,9,10,11,12,13,14,15);
}

// ---------------------------------------------------------------------------
// CDNA5 async global->LDS DMA (bypasses VGPRs, tracked by ASYNCcnt).
// GVS addressing: mem_addr = SGPR64 + VGPR32 byte offset.
// lds_off is the 32-bit LDS byte offset (low 32 bits of the generic pointer).
// ---------------------------------------------------------------------------
__device__ __forceinline__ void async_load_b128(uint32_t lds_off,
                                                const void* saddr,
                                                uint32_t goff) {
  asm volatile("global_load_async_to_lds_b128 %0, %1, %2"
               :: "v"(lds_off), "v"(goff), "s"(saddr) : "memory");
}
__device__ __forceinline__ void wait_asynccnt_4() {
  asm volatile("s_wait_asynccnt 0x4" ::: "memory");
}
__device__ __forceinline__ void wait_asynccnt_0() {
  asm volatile("s_wait_asynccnt 0x0" ::: "memory");
}

// ---------------------------------------------------------------------------
// f32 -> bf16 conversion (vectorized 4-wide; all sizes are multiples of 4)
// ---------------------------------------------------------------------------
__global__ void cvt_f32_bf16(const float* __restrict__ in,
                             bf16* __restrict__ out, int n4) {
  int i = blockIdx.x * blockDim.x + threadIdx.x;
  if (i >= n4) return;
  float4 v = ((const float4*)in)[i];
  v4bf o;
  o[0] = f2bf(v.x); o[1] = f2bf(v.y); o[2] = f2bf(v.z); o[3] = f2bf(v.w);
  ((v4bf*)out)[i] = o;
}

// ---------------------------------------------------------------------------
// Tiled WMMA GEMM:  C[M,N] = A[M,K](bf16) * Bt[N,K](bf16)^T + bias[N]
// Block tile 128(M) x 128(N), 8 waves -> each wave 32x64 (8 accumulators).
// K staged in chunks of 32 through double-buffered LDS filled with
// global_load_async_to_lds_b128 (ASYNCcnt-fenced). The steady-state loop is
// branch-free (always prefetch next slab + wait<=4); the last chunk is
// peeled so no conditionals remain inside the hot loop.
// ---------------------------------------------------------------------------
__global__ __launch_bounds__(256)
void gemm_bf16_wmma(const bf16* __restrict__ A, const bf16* __restrict__ Bt,
                    const float* __restrict__ bias, float* __restrict__ C,
                    int M, int N, int K) {
  __shared__ __align__(128) bf16 ldsA[2][128 * 32];   // [buf][row][k]
  __shared__ __align__(128) bf16 ldsB[2][128 * 32];   // [buf][n][k]

  const int tid  = threadIdx.x;
  const int lane = tid & 31;
  const int w    = tid >> 5;
  const int wm   = w >> 1;          // 0..3 -> 32-row M subtile
  const int wn   = w & 1;           // 0..1 -> 64-col N subtile
  const int g    = lane >> 4;       // lane k-group
  const int ln   = lane & 15;

  const int mBase = blockIdx.y * 128;
  const int nBase = blockIdx.x * 128;

  // per-thread async-copy assignment: 2 x b128 for A, 2 x b128 for B per chunk
  const int c0r = tid >> 2, c0c = (tid & 3) * 8;           // chunk tid
  const int c1r = (tid + 256) >> 2, c1c = ((tid + 256) & 3) * 8;

  auto stage = [&](int buf, int kk) {
    async_load_b128((uint32_t)(size_t)&ldsA[buf][c0r * 32 + c0c], A,
                    (uint32_t)(((size_t)(mBase + c0r) * K + kk + c0c) * 2));
    async_load_b128((uint32_t)(size_t)&ldsA[buf][c1r * 32 + c1c], A,
                    (uint32_t)(((size_t)(mBase + c1r) * K + kk + c1c) * 2));
    async_load_b128((uint32_t)(size_t)&ldsB[buf][c0r * 32 + c0c], Bt,
                    (uint32_t)(((size_t)(nBase + c0r) * K + kk + c0c) * 2));
    async_load_b128((uint32_t)(size_t)&ldsB[buf][c1r * 32 + c1c], Bt,
                    (uint32_t)(((size_t)(nBase + c1r) * K + kk + c1c) * 2));
  };

  v8f acc[2][4] = {};

  auto compute = [&](int buf) {
    const bf16* bufA = ldsA[buf];
    const bf16* bufB = ldsB[buf];
    v16bf a[2];
#pragma unroll
    for (int ms = 0; ms < 2; ++ms) {
      const int arow = (wm * 32 + ms * 16 + ln) * 32;
      a[ms] = cat8(*(const v8bf*)&bufA[arow + g * 8],
                   *(const v8bf*)&bufA[arow + 16 + g * 8]);
    }
#pragma unroll
    for (int t = 0; t < 4; ++t) {
      v16bf b = cat8(*(const v8bf*)&bufB[(wn * 64 + t * 16 + ln) * 32 + g * 8],
                     *(const v8bf*)&bufB[(wn * 64 + t * 16 + ln) * 32 + 16 + g * 8]);
#pragma unroll
      for (int ms = 0; ms < 2; ++ms)
        acc[ms][t] = __builtin_amdgcn_wmma_f32_16x16x32_bf16(
            false, a[ms], false, b, (short)0, acc[ms][t], false, false);
    }
  };

  stage(0, 0);                                   // preload first K-chunk
  const int last = K - 32;
  int kk = 0;
  for (; kk < last; kk += 32) {                  // branch-free steady state
    const int cur = (kk >> 5) & 1;
    stage(cur ^ 1, kk + 32);                     // prefetch next chunk
    wait_asynccnt_4();                           // in-order => current landed
    __syncthreads();                             // make all waves' DMA visible
    compute(cur);
    __syncthreads();                             // before buf is re-filled
  }
  {                                              // peeled last chunk
    const int cur = (kk >> 5) & 1;
    wait_asynccnt_0();
    __syncthreads();
    compute(cur);
  }

#pragma unroll
  for (int ms = 0; ms < 2; ++ms) {
#pragma unroll
    for (int t = 0; t < 4; ++t) {
      const int col = nBase + wn * 64 + t * 16 + ln;
      const float bv = bias ? bias[col] : 0.0f;
#pragma unroll
      for (int i = 0; i < 8; ++i) {
        const int row = mBase + wm * 32 + ms * 16 + g * 8 + i;
        C[(size_t)row * N + col] = acc[ms][t][i] + bv;
      }
    }
  }
}

// ---------------------------------------------------------------------------
// Fused per-head LayerNorm + RoPE, f32 in -> bf16 out.
// One wave32 per 128-wide head row; lane t owns d = {t, t+32, t+64, t+96}
// so RoPE pairs (d, d+64) stay lane-local.
// ---------------------------------------------------------------------------
__global__ __launch_bounds__(256)
void ln_rope_kernel(const float* __restrict__ in, bf16* __restrict__ out,
                    const float* __restrict__ gamma, const float* __restrict__ beta,
                    int nheads, int nrows) {
  const int row = blockIdx.x * 8 + (threadIdx.x >> 5);
  if (row >= nrows) return;
  const int lane = threadIdx.x & 31;
  const float* p = in + (size_t)row * HD;

  float x0 = p[lane], x1 = p[lane + 32], x2 = p[lane + 64], x3 = p[lane + 96];

  float s = x0 + x1 + x2 + x3;
#pragma unroll
  for (int off = 1; off < 32; off <<= 1) s += __shfl_xor(s, off);
  const float mu = s * (1.0f / 128.0f);

  const float d0 = x0 - mu, d1 = x1 - mu, d2 = x2 - mu, d3 = x3 - mu;
  float vs = d0 * d0 + d1 * d1 + d2 * d2 + d3 * d3;
#pragma unroll
  for (int off = 1; off < 32; off <<= 1) vs += __shfl_xor(vs, off);
  const float rinv = rsqrtf(vs * (1.0f / 128.0f) + EPS);

  const float y0 = d0 * rinv * gamma[lane]      + beta[lane];
  const float y1 = d1 * rinv * gamma[lane + 32] + beta[lane + 32];
  const float y2 = d2 * rinv * gamma[lane + 64] + beta[lane + 64];
  const float y3 = d3 * rinv * gamma[lane + 96] + beta[lane + 96];

  // RoPE: freq index = d % 64; pair (d, d+64)
  const float pos = (float)(row / nheads);
  const float f0 = pos * __expf(-(float)(2 * lane)        * (1.0f / 128.0f) * LOG_ROPE_THETA);
  const float f1 = pos * __expf(-(float)(2 * (lane + 32)) * (1.0f / 128.0f) * LOG_ROPE_THETA);
  const float c0 = cosf(f0), s0 = sinf(f0);
  const float c1 = cosf(f1), s1 = sinf(f1);

  bf16* o = out + (size_t)row * HD;
  o[lane]      = f2bf(y0 * c0 - y2 * s0);
  o[lane + 32] = f2bf(y1 * c1 - y3 * s1);
  o[lane + 64] = f2bf(y2 * c0 + y0 * s0);
  o[lane + 96] = f2bf(y3 * c1 + y1 * s1);
}

// ---------------------------------------------------------------------------
// Flash attention (causal, GQA 4:1), all matmuls on WMMA bf16.
// Grid: (S/128, NH). Block 256 = 8 waves; wave w owns query rows
// [qs + 16w, qs + 16w + 15]. Key blocks of 64.
// ---------------------------------------------------------------------------
__global__ __launch_bounds__(256)
void attn_wmma_kernel(const bf16* __restrict__ Qb, const bf16* __restrict__ Kb,
                      const bf16* __restrict__ Vb, bf16* __restrict__ Ctx) {
  __shared__ __align__(128) bf16 Vt[HD * 64];       // V tile transposed: [d][key]
  __shared__ __align__(128) bf16 Pw[8][16 * 64];    // wave-private P tiles

  const int tid  = threadIdx.x;
  const int lane = tid & 31;
  const int w    = tid >> 5;
  const int g    = lane >> 4;
  const int ln   = lane & 15;

  const int h  = blockIdx.y;      // query head
  const int kh = h >> 2;          // kv head (N_REP = 4)
  const int qs = blockIdx.x * 128;
  const int qrow = qs + w * 16 + ln;     // this lane's A-fragment row

  // Q fragments for the wave's 16x128 strip: 4 chunks of K=32 over head dim
  v16bf aQ[4];
  {
    const bf16* qp = Qb + (size_t)qrow * DQ + h * HD;
#pragma unroll
    for (int c = 0; c < 4; ++c)
      aQ[c] = cat8(*(const v8bf*)&qp[c * 32 + g * 8],
                   *(const v8bf*)&qp[c * 32 + 16 + g * 8]);
  }

  v8f O[8] = {};
  float mrun[8], lrun[8];
#pragma unroll
  for (int i = 0; i < 8; ++i) { mrun[i] = -1e30f; lrun[i] = 0.0f; }

  for (int ks = 0; ks < qs + 128; ks += 64) {
    // ---- stage V tile (64 keys x 128 d) transposed into LDS ----
#pragma unroll
    for (int it = 0; it < 4; ++it) {
      const int idx = tid + it * 256;          // 0..1023
      const int key = idx >> 4, dc = idx & 15; // 8 d-elements per load
      v8bf vv = *(const v8bf*)&Vb[(size_t)(ks + key) * DKV + kh * HD + dc * 8];
#pragma unroll
      for (int j = 0; j < 8; ++j) Vt[(dc * 8 + j) * 64 + key] = vv[j];
    }
    __syncthreads();

    // ---- scores: S = Q (16x128) x K^T (128x64), 16 WMMAs ----
    v8f sacc[4] = {};
#pragma unroll
    for (int t = 0; t < 4; ++t) {
      const bf16* kp = Kb + (size_t)(ks + t * 16 + ln) * DKV + kh * HD;
#pragma unroll
      for (int c = 0; c < 4; ++c) {
        v16bf bK = cat8(*(const v8bf*)&kp[c * 32 + g * 8],
                        *(const v8bf*)&kp[c * 32 + 16 + g * 8]);
        sacc[t] = __builtin_amdgcn_wmma_f32_16x16x32_bf16(
            false, aQ[c], false, bK, (short)0, sacc[t], false, false);
      }
    }

    // ---- scale + causal mask ----
#pragma unroll
    for (int t = 0; t < 4; ++t) {
      const int kcol = ks + t * 16 + ln;
#pragma unroll
      for (int i = 0; i < 8; ++i) {
        const int q = qs + w * 16 + g * 8 + i;
        const float sv = sacc[t][i] * SOFTMAX_SCALE;
        sacc[t][i] = (kcol <= q) ? sv : -1e30f;
      }
    }

    // ---- online softmax: row max, rescale O and l ----
    float mnew[8];
#pragma unroll
    for (int i = 0; i < 8; ++i) {
      float m = fmaxf(fmaxf(sacc[0][i], sacc[1][i]),
                      fmaxf(sacc[2][i], sacc[3][i]));
#pragma unroll
      for (int off = 1; off < 16; off <<= 1) m = fmaxf(m, __shfl_xor(m, off));
      mnew[i] = fmaxf(mrun[i], m);
      const float rs = __expf(mrun[i] - mnew[i]);
      lrun[i] *= rs;
#pragma unroll
      for (int t = 0; t < 8; ++t) O[t][i] *= rs;
      mrun[i] = mnew[i];
    }

    // ---- P = exp(S - m): store to wave-private LDS in row-major [16][64] ----
#pragma unroll
    for (int t = 0; t < 4; ++t) {
#pragma unroll
      for (int i = 0; i < 8; ++i) {
        const float p = __expf(sacc[t][i] - mnew[i]);
        sacc[t][i] = p;
        Pw[w][(g * 8 + i) * 64 + t * 16 + ln] = f2bf(p);
      }
    }
#pragma unroll
    for (int i = 0; i < 8; ++i) {
      float rsum = sacc[0][i] + sacc[1][i] + sacc[2][i] + sacc[3][i];
#pragma unroll
      for (int off = 1; off < 16; off <<= 1) rsum += __shfl_xor(rsum, off);
      lrun[i] += rsum;
    }

    // ---- re-fragment P as A matrix (two 16x32 chunks over keys) ----
    v16bf aP[2];
#pragma unroll
    for (int c = 0; c < 2; ++c)
      aP[c] = cat8(*(const v8bf*)&Pw[w][ln * 64 + c * 32 + g * 8],
                   *(const v8bf*)&Pw[w][ln * 64 + c * 32 + 16 + g * 8]);

    // ---- O += P (16x64) x V (64x128), 16 WMMAs ----
#pragma unroll
    for (int t = 0; t < 8; ++t) {
#pragma unroll
      for (int c = 0; c < 2; ++c) {
        v16bf bV = cat8(*(const v8bf*)&Vt[(t * 16 + ln) * 64 + c * 32 + g * 8],
                        *(const v8bf*)&Vt[(t * 16 + ln) * 64 + c * 32 + 16 + g * 8]);
        O[t] = __builtin_amdgcn_wmma_f32_16x16x32_bf16(
            false, aP[c], false, bV, (short)0, O[t], false, false);
      }
    }
    __syncthreads();   // before Vt is overwritten next key block
  }

  // ---- finalize: O / l, write bf16 context (feeds the O-projection GEMM) ----
#pragma unroll
  for (int t = 0; t < 8; ++t) {
#pragma unroll
    for (int i = 0; i < 8; ++i) {
      const int q = qs + w * 16 + g * 8 + i;
      Ctx[(size_t)q * DQ + h * HD + t * 16 + ln] = f2bf(O[t][i] / lrun[i]);
    }
  }
}

// ---------------------------------------------------------------------------
// Host-side orchestration
// ---------------------------------------------------------------------------
extern "C" void kernel_launch(void* const* d_in, const int* in_sizes, int n_in,
                              void* d_out, int out_size, void* d_ws, size_t ws_size,
                              hipStream_t stream) {
  const float* hidden = (const float*)d_in[0];
  const float* Wq = (const float*)d_in[1];  const float* bq = (const float*)d_in[2];
  const float* Wk = (const float*)d_in[3];  const float* bk = (const float*)d_in[4];
  const float* Wv = (const float*)d_in[5];  const float* bv = (const float*)d_in[6];
  const float* Wo = (const float*)d_in[7];  const float* bo = (const float*)d_in[8];
  const float* qn_g = (const float*)d_in[9];  const float* qn_b = (const float*)d_in[10];
  const float* kn_g = (const float*)d_in[11]; const float* kn_b = (const float*)d_in[12];
  float* out = (float*)d_out;

  // workspace layout (bytes)
  char* ws = (char*)d_ws;
  size_t off = 0;
  auto alloc = [&](size_t bytes) { char* p = ws + off; off += (bytes + 255) & ~(size_t)255; return p; };
  bf16* Xb   = (bf16*)alloc((size_t)S * HID * 2);     // 16.8 MB
  bf16* Wqb  = (bf16*)alloc((size_t)DQ * HID * 2);    // 33.6 MB
  bf16* Wkb  = (bf16*)alloc((size_t)DKV * HID * 2);   //  8.4 MB
  bf16* Wvb  = (bf16*)alloc((size_t)DKV * HID * 2);   //  8.4 MB
  bf16* Wob  = (bf16*)alloc((size_t)HID * HID * 2);   // 33.6 MB
  float* Qf  = (float*)alloc((size_t)S * DQ * 4);     // 33.6 MB
  float* Kf  = (float*)alloc((size_t)S * DKV * 4);    //  8.4 MB
  float* Vf  = (float*)alloc((size_t)S * DKV * 4);    //  8.4 MB
  bf16* Qb   = (bf16*)alloc((size_t)S * DQ * 2);      // 16.8 MB
  bf16* Kb   = (bf16*)alloc((size_t)S * DKV * 2);     //  4.2 MB
  bf16* Vb   = (bf16*)alloc((size_t)S * DKV * 2);     //  4.2 MB
  bf16* Ctxb = (bf16*)alloc((size_t)S * DQ * 2);      // 16.8 MB
  (void)ws_size;

  const int BLK = 256;
  auto cvt = [&](const float* src, bf16* dst, size_t n) {
    int n4 = (int)(n / 4);
    cvt_f32_bf16<<<(n4 + BLK - 1) / BLK, BLK, 0, stream>>>(src, dst, n4);
  };

  // 1) bf16 copies of activations and weights
  cvt(hidden, Xb, (size_t)S * HID);
  cvt(Wq, Wqb, (size_t)DQ * HID);
  cvt(Wk, Wkb, (size_t)DKV * HID);
  cvt(Wv, Wvb, (size_t)DKV * HID);
  cvt(Wo, Wob, (size_t)HID * HID);

  // 2) QKV projections (WMMA GEMM, async-DMA staged, double-buffered)
  gemm_bf16_wmma<<<dim3(DQ / 128, S / 128), BLK, 0, stream>>>(Xb, Wqb, bq, Qf, S, DQ, HID);
  gemm_bf16_wmma<<<dim3(DKV / 128, S / 128), BLK, 0, stream>>>(Xb, Wkb, bk, Kf, S, DKV, HID);
  gemm_bf16_wmma<<<dim3(DKV / 128, S / 128), BLK, 0, stream>>>(Xb, Wvb, bv, Vf, S, DKV, HID);

  // 3) per-head LayerNorm + RoPE -> bf16 Q/K; V just converted
  ln_rope_kernel<<<(S * NH) / 8, BLK, 0, stream>>>(Qf, Qb, qn_g, qn_b, NH, S * NH);
  ln_rope_kernel<<<(S * NKV) / 8, BLK, 0, stream>>>(Kf, Kb, kn_g, kn_b, NKV, S * NKV);
  cvt(Vf, Vb, (size_t)S * DKV);

  // 4) causal GQA flash attention (WMMA)
  attn_wmma_kernel<<<dim3(S / 128, NH), BLK, 0, stream>>>(Qb, Kb, Vb, Ctxb);

  // 5) output projection (WMMA GEMM) -> f32 result
  gemm_bf16_wmma<<<dim3(HID / 128, S / 128), BLK, 0, stream>>>(Ctxb, Wob, bo, out, S, HID, HID);
}
